// mLSTMBlock_44684839747979
// MI455X (gfx1250) — matrix operand order repaired and checked by reading, compiled, not run on hardware
//
#include <hip/hip_runtime.h>
#include <hip/hip_bf16.h>
#include <stdint.h>

typedef __attribute__((ext_vector_type(16))) _Float16 v16h;
typedef __attribute__((ext_vector_type(8)))  _Float16 v8h;
typedef __attribute__((ext_vector_type(8)))  float    v8f;

#define Bsz  8192
#define INF  1024
#define HIDF 2048
#define NH   8
#define UPF  2048
#define HSF  256
#define EPSV 1e-5f

// ---------------- CDNA5 async global->LDS (ASYNCcnt-tracked DMA) ----------------
__device__ __forceinline__ void g2l_b128(const _Float16* g, const _Float16* l) {
  unsigned lds = (unsigned)(uintptr_t)l;   // low 32 bits of generic shared ptr = LDS byte addr
  asm volatile("global_load_async_to_lds_b128 %0, %1, off"
               :: "v"(lds), "v"(g) : "memory");
}
#define WAIT_ASYNC_4() asm volatile("s_wait_asynccnt 0x4" ::: "memory")
#define WAIT_ASYNC_2() asm volatile("s_wait_asynccnt 0x2" ::: "memory")
#define WAIT_ASYNC_0() asm volatile("s_wait_asynccnt 0x0" ::: "memory")

// ---------------- LayerNorm -> f16 ----------------
__global__ __launch_bounds__(256) void ln_k(const float* __restrict__ x,
                                            const float* __restrict__ g,
                                            const float* __restrict__ b,
                                            _Float16* __restrict__ xn_h) {
  __shared__ float r1[256], r2[256];
  int row = blockIdx.x;
  const float* xr = x + (size_t)row * INF;
  float v[4]; float s = 0.f, s2 = 0.f;
#pragma unroll
  for (int i = 0; i < 4; ++i) {
    float t = xr[threadIdx.x + i * 256];
    v[i] = t; s += t; s2 += t * t;
  }
  r1[threadIdx.x] = s; r2[threadIdx.x] = s2; __syncthreads();
  for (int off = 128; off > 0; off >>= 1) {
    if (threadIdx.x < off) { r1[threadIdx.x] += r1[threadIdx.x + off]; r2[threadIdx.x] += r2[threadIdx.x + off]; }
    __syncthreads();
  }
  float mean = r1[0] / INF;
  float var  = r2[0] / INF - mean * mean;
  float rstd = rsqrtf(var + EPSV);
#pragma unroll
  for (int i = 0; i < 4; ++i) {
    int c = threadIdx.x + i * 256;
    xn_h[(size_t)row * INF + c] = (_Float16)(((v[i] - mean) * rstd) * g[c] + b[c]);
  }
}

// ---------------- Transpose f32 [R x C] -> f16 [C x R] (batched) ----------------
__global__ void tr_f32_f16_k(const float* __restrict__ src, _Float16* __restrict__ dst,
                             int R, int C, long long sStride, long long dStride) {
  __shared__ float t[32][33];
  src += (size_t)blockIdx.z * sStride;
  dst += (size_t)blockIdx.z * dStride;
  int r0 = blockIdx.y * 32, c0 = blockIdx.x * 32;
  for (int i = threadIdx.y; i < 32; i += 8)
    t[i][threadIdx.x] = src[(size_t)(r0 + i) * C + c0 + threadIdx.x];
  __syncthreads();
  for (int i = threadIdx.y; i < 32; i += 8)
    dst[(size_t)(c0 + i) * R + r0 + threadIdx.x] = (_Float16)t[threadIdx.x][i];
}

// ---------------- WMMA fragment gathers from LDS ----------------
#define LDS_K 40   // 32 + 8 pad halves; 80B row stride (16B aligned)

__device__ __forceinline__ v16h fragA(const _Float16* s, int row, int half) {
  // A 16x32 f16: lane<16 -> K{0..7,16..23}; lane>=16 -> K{8..15,24..31}
  v8h lo = *(const v8h*)(s + row * LDS_K + half * 8);
  v8h hi = *(const v8h*)(s + row * LDS_K + 16 + half * 8);
  v16h f;
#pragma unroll
  for (int e = 0; e < 8; ++e) { f[e] = lo[e]; f[8 + e] = hi[e]; }
  return f;
}
__device__ __forceinline__ v16h fragB(const _Float16* s, int col, int half) {
  // B 32x16 f16 (stored as B^T, K contiguous): lane<16 -> K{0..7,16..23}; lane>=16 -> K{8..15,24..31}
  v8h lo = *(const v8h*)(s + col * LDS_K + half * 8);
  v8h hi = *(const v8h*)(s + col * LDS_K + 16 + half * 8);
  v16h f;
#pragma unroll
  for (int e = 0; e < 8; ++e) { f[e] = lo[e]; f[8 + e] = hi[e]; }
  return f;
}

// ---------------- Generic WMMA GEMM:  C = alpha*(A@B + bias) [+ D]  ----------------
// A: [M x K] f16 row-major (lda). BT: [N x K] f16 row-major (ldb).
// Tile 128(M) x 128(N), K-step 32. 8 waves; wave -> 32x64 (2x4 WMMA, 8 per K-step).
// Double-buffered LDS fed by async global->LDS DMA, last iteration peeled.
#define GTM 128
#define GTN 128
#define GKT 32
__global__ __launch_bounds__(256) void gemm_wmma_k(
    const _Float16* __restrict__ A, int lda, long long strideA,
    const _Float16* __restrict__ BT, int ldb, long long strideB,
    float* __restrict__ C, int ldc, long long strideC,
    const float* __restrict__ bias, long long strideBias,
    const float* __restrict__ D, int ldd,
    _Float16* __restrict__ Ch, int ldch,
    float alpha, int K) {
  __shared__ _Float16 As[2][GTM * LDS_K];
  __shared__ _Float16 Bs[2][GTN * LDS_K];
  A    += (size_t)blockIdx.z * strideA;
  BT   += (size_t)blockIdx.z * strideB;
  C    += (size_t)blockIdx.z * strideC;
  bias += (size_t)blockIdx.z * strideBias;
  int m0 = blockIdx.y * GTM, n0 = blockIdx.x * GTN;
  int tid = threadIdx.x, lane = tid & 31, wave = tid >> 5;
  int wm = wave >> 1, wn = wave & 1;       // 4 x 2 waves -> 32x64 per wave
  int half = lane >> 4, r = lane & 15;

  // per-thread staging: 16B units; A: 2 units, B: 2 units (128 rows x 4 segs each)
  int ra = tid >> 2;            // 0..63
  int sa = (tid & 3) * 8;       // half offset within 32-wide K slab
  const _Float16* aSrc0 = A  + (size_t)(m0 + ra)      * lda + sa;
  const _Float16* aSrc1 = A  + (size_t)(m0 + ra + 64) * lda + sa;
  const _Float16* bSrc0 = BT + (size_t)(n0 + ra)      * ldb + sa;
  const _Float16* bSrc1 = BT + (size_t)(n0 + ra + 64) * ldb + sa;
  int off0 = ra * LDS_K + sa, off1 = (ra + 64) * LDS_K + sa;

  v8f acc[2][4] = {};
  int nIter = K / GKT;
  // prologue: tile 0 -> buffer 0
  g2l_b128(aSrc0, &As[0][off0]);
  g2l_b128(aSrc1, &As[0][off1]);
  g2l_b128(bSrc0, &Bs[0][off0]);
  g2l_b128(bSrc1, &Bs[0][off1]);
  // steady state: no tail branch inside the loop
  for (int it = 0; it < nIter - 1; ++it) {
    int cur = it & 1, nxt = cur ^ 1;
    size_t ko = (size_t)(it + 1) * GKT;
    g2l_b128(aSrc0 + ko, &As[nxt][off0]);
    g2l_b128(aSrc1 + ko, &As[nxt][off1]);
    g2l_b128(bSrc0 + ko, &Bs[nxt][off0]);
    g2l_b128(bSrc1 + ko, &Bs[nxt][off1]);
    WAIT_ASYNC_4();            // tile `it` landed; newest 4 still in flight
    __syncthreads();
    v16h af[2], bf[4];
#pragma unroll
    for (int i = 0; i < 2; ++i) af[i] = fragA(&As[cur][0], wm * 32 + i * 16 + r, half);
#pragma unroll
    for (int j = 0; j < 4; ++j) bf[j] = fragB(&Bs[cur][0], wn * 64 + j * 16 + r, half);
#pragma unroll
    for (int i = 0; i < 2; ++i)
#pragma unroll
      for (int j = 0; j < 4; ++j)
        acc[i][j] = __builtin_amdgcn_wmma_f32_16x16x32_f16(
            false, af[i], false, bf[j], (short)0, acc[i][j], false, false);
    __syncthreads();           // readers done before this buffer is refilled (it+2)
  }
  // peeled final iteration
  {
    int cur = (nIter - 1) & 1;
    WAIT_ASYNC_0();
    __syncthreads();
    v16h af[2], bf[4];
#pragma unroll
    for (int i = 0; i < 2; ++i) af[i] = fragA(&As[cur][0], wm * 32 + i * 16 + r, half);
#pragma unroll
    for (int j = 0; j < 4; ++j) bf[j] = fragB(&Bs[cur][0], wn * 64 + j * 16 + r, half);
#pragma unroll
    for (int i = 0; i < 2; ++i)
#pragma unroll
      for (int j = 0; j < 4; ++j)
        acc[i][j] = __builtin_amdgcn_wmma_f32_16x16x32_f16(
            false, af[i], false, bf[j], (short)0, acc[i][j], false, false);
  }
#pragma unroll
  for (int i = 0; i < 2; ++i)
#pragma unroll
    for (int j = 0; j < 4; ++j) {
      int rowBase = m0 + wm * 32 + i * 16;
      int n = n0 + wn * 64 + j * 16 + r;
      float bn = bias ? bias[n] : 0.f;
#pragma unroll
      for (int jj = 0; jj < 8; ++jj) {
        int m = rowBase + jj + 8 * half;   // C layout: VGPR jj -> M=jj (+8 for lanes 16-31)
        float val = alpha * (acc[i][j][jj] + bn);
        if (D)  val += D[(size_t)m * ldd + n];
        C[(size_t)m * ldc + n] = val;
        if (Ch) Ch[(size_t)m * ldch + n] = (_Float16)val;
      }
    }
}

// ---------------- denom = rowmax |n_t^T @ q| via WMMA, K = 8192 ----------------
#define DTM 64
#define DTN 64
__global__ __launch_bounds__(256) void denom_wmma_k(const _Float16* __restrict__ nT,
                                                    const _Float16* __restrict__ qT,
                                                    float* __restrict__ denom, int K) {
  __shared__ _Float16 As[2][DTM * LDS_K];
  __shared__ _Float16 Bs[2][DTN * LDS_K];
  __shared__ float red[DTM][DTN + 1];
  int m0 = blockIdx.y * DTM, n0 = blockIdx.x * DTN;
  int tid = threadIdx.x, lane = tid & 31, wave = tid >> 5;
  int wm = wave >> 2, wn = wave & 3;         // 2 x 4 wave grid, wave -> 32x16
  int half = lane >> 4, r = lane & 15;

  int ra = tid >> 2;            // 0..63
  int sa = (tid & 3) * 8;
  const _Float16* aSrc = nT + (size_t)(m0 + ra) * K + sa;
  const _Float16* bSrc = qT + (size_t)(n0 + ra) * K + sa;
  int off = ra * LDS_K + sa;

  v8f acc[2] = {};
  int nIter = K / GKT;
  g2l_b128(aSrc, &As[0][off]);
  g2l_b128(bSrc, &Bs[0][off]);
  for (int it = 0; it < nIter - 1; ++it) {
    int cur = it & 1, nxt = cur ^ 1;
    size_t ko = (size_t)(it + 1) * GKT;
    g2l_b128(aSrc + ko, &As[nxt][off]);
    g2l_b128(bSrc + ko, &Bs[nxt][off]);
    WAIT_ASYNC_2();
    __syncthreads();
    v16h bf = fragB(&Bs[cur][0], wn * 16 + r, half);
#pragma unroll
    for (int i = 0; i < 2; ++i) {
      v16h af = fragA(&As[cur][0], wm * 32 + i * 16 + r, half);
      acc[i] = __builtin_amdgcn_wmma_f32_16x16x32_f16(false, af, false, bf, (short)0, acc[i], false, false);
    }
    __syncthreads();
  }
  {
    int cur = (nIter - 1) & 1;
    WAIT_ASYNC_0();
    __syncthreads();
    v16h bf = fragB(&Bs[cur][0], wn * 16 + r, half);
#pragma unroll
    for (int i = 0; i < 2; ++i) {
      v16h af = fragA(&As[cur][0], wm * 32 + i * 16 + r, half);
      acc[i] = __builtin_amdgcn_wmma_f32_16x16x32_f16(false, af, false, bf, (short)0, acc[i], false, false);
    }
    __syncthreads();
  }
#pragma unroll
  for (int i = 0; i < 2; ++i) {
    int rowBase = wm * 32 + i * 16;
    int n = wn * 16 + r;
#pragma unroll
    for (int jj = 0; jj < 8; ++jj) red[rowBase + jj + 8 * half][n] = fabsf(acc[i][jj]);
  }
  __syncthreads();
  if (tid < DTM) {
    float mx = 0.f;
    for (int c = 0; c < DTN; ++c) mx = fmaxf(mx, red[tid][c]);
    atomicMax((int*)&denom[m0 + tid], __float_as_int(mx));  // all values >= 0
  }
}

__global__ void denom_init_k(float* d) { d[blockIdx.x * 256 + threadIdx.x] = 0.f; }

// ---------------- causal conv1d(k=4) + SiLU -> f16 ----------------
__global__ __launch_bounds__(256) void conv_silu_k(const float* __restrict__ xul,
                                                   const float* __restrict__ cw,
                                                   const float* __restrict__ cb,
                                                   _Float16* __restrict__ xc_h) {
  size_t idx = (size_t)blockIdx.x * 256 + threadIdx.x;  // over B*UP
  int t = (int)(idx % UPF);
  size_t base = idx - t;
  float acc = cb[0];
#pragma unroll
  for (int j = 0; j < 4; ++j) {
    int s = t + j - 3;
    if (s >= 0) acc += cw[j] * xul[base + s];
  }
  xc_h[idx] = (_Float16)(acc / (1.f + expf(-acc)));
}

// ---------------- mLSTM state update (writes m_t, c_t, n_t to d_out) ----------------
__global__ __launch_bounds__(256) void state_k(const float* __restrict__ i_t, const float* __restrict__ f_t,
                                               const float* __restrict__ kbuf, const float* __restrict__ vbuf,
                                               const float* __restrict__ c_prev, const float* __restrict__ n_prev,
                                               const float* __restrict__ m_prev,
                                               float* __restrict__ m_out, float* __restrict__ c_out,
                                               float* __restrict__ n_out) {
  size_t idx = (size_t)blockIdx.x * 256 + threadIdx.x;
  float it = i_t[idx];
  float fp = f_t[idx] + m_prev[idx];
  float mt = fmaxf(fp, it);
  float ie = expf(it - mt), fe = expf(fp - mt);
  float kk = kbuf[idx];
  c_out[idx] = fe * c_prev[idx] + ie * (vbuf[idx] * kk);
  n_out[idx] = fe * n_prev[idx] + ie * kk;
  m_out[idx] = mt;
}

// ---------------- h_t + GroupNorm + gated skip -> pre (f16), h_t (out) ----------------
__global__ __launch_bounds__(256) void head_out_k(const float* __restrict__ o_pre,
                                                  const float* __restrict__ c_t, const float* __restrict__ q,
                                                  const float* __restrict__ denom,
                                                  const float* __restrict__ gn_g, const float* __restrict__ gn_b,
                                                  const float* __restrict__ x_skip, const float* __restrict__ xur,
                                                  float* __restrict__ h_out, _Float16* __restrict__ pre_h) {
  __shared__ float r1[256], r2[256];
  int b = blockIdx.x, h = blockIdx.y, t = threadIdx.x;
  size_t idx = (size_t)b * HIDF + h * HSF + t;
  int g = h * HSF + t;
  float o  = 1.f / (1.f + expf(-o_pre[idx]));
  float hv = o * c_t[idx] * q[idx] / denom[g];
  h_out[idx] = hv;
  r1[t] = hv; r2[t] = hv * hv; __syncthreads();
  for (int off = 128; off > 0; off >>= 1) {
    if (t < off) { r1[t] += r1[t + off]; r2[t] += r2[t + off]; }
    __syncthreads();
  }
  float mean = r1[0] / HSF;
  float var  = r2[0] / HSF - mean * mean;
  float hn = (hv - mean) * rsqrtf(var + EPSV) * gn_g[g] + gn_b[g];
  float xu = xur[idx];
  pre_h[idx] = (_Float16)((hn + x_skip[idx]) * (xu / (1.f + expf(-xu))));
}

// ==================================================================================
static inline char* wsAlloc(char*& p, size_t bytes) {
  char* r = p; p += (bytes + 255) & ~(size_t)255; return r;
}

extern "C" void kernel_launch(void* const* d_in, const int* in_sizes, int n_in,
                              void* d_out, int out_size, void* d_ws, size_t ws_size,
                              hipStream_t stream) {
  (void)in_sizes; (void)n_in; (void)out_size; (void)ws_size;
  const float* x      = (const float*)d_in[0];
  const float* c_prev = (const float*)d_in[2];
  const float* n_prev = (const float*)d_in[3];
  const float* m_prev = (const float*)d_in[4];
  const float* ln_g   = (const float*)d_in[5];
  const float* ln_b   = (const float*)d_in[6];
  const float* w_ul   = (const float*)d_in[7];
  const float* b_ul   = (const float*)d_in[8];
  const float* w_ur   = (const float*)d_in[9];
  const float* b_ur   = (const float*)d_in[10];
  const float* conv_w = (const float*)d_in[11];
  const float* conv_b = (const float*)d_in[12];
  const float* w_skip = (const float*)d_in[13];
  const float* b_skip = (const float*)d_in[14];
  const float* w_q    = (const float*)d_in[15];
  const float* b_q    = (const float*)d_in[16];
  const float* w_k    = (const float*)d_in[17];
  const float* b_k    = (const float*)d_in[18];
  const float* w_v    = (const float*)d_in[19];
  const float* b_v    = (const float*)d_in[20];
  const float* w_i    = (const float*)d_in[21];
  const float* b_i    = (const float*)d_in[22];
  const float* w_f    = (const float*)d_in[23];
  const float* b_f    = (const float*)d_in[24];
  const float* w_o    = (const float*)d_in[25];
  const float* b_o    = (const float*)d_in[26];
  const float* gn_g   = (const float*)d_in[27];
  const float* gn_b   = (const float*)d_in[28];
  const float* w_down = (const float*)d_in[29];
  const float* b_down = (const float*)d_in[30];

  float* out = (float*)d_out;
  const size_t o1 = (size_t)Bsz * INF;            // h_t
  const size_t o2 = o1 + (size_t)Bsz * HIDF;      // c_t
  const size_t o3 = o2 + (size_t)Bsz * HIDF;      // n_t
  const size_t o4 = o3 + (size_t)Bsz * HIDF;      // m_t

  char* p = (char*)d_ws;
  _Float16* xn_h    = (_Float16*)wsAlloc(p, (size_t)Bsz * INF * 2);
  _Float16* wulT    = (_Float16*)wsAlloc(p, (size_t)UPF * INF * 2);
  _Float16* wurT    = (_Float16*)wsAlloc(p, (size_t)HIDF * INF * 2);
  _Float16* wskipT  = (_Float16*)wsAlloc(p, (size_t)HIDF * UPF * 2);
  _Float16* wiT     = (_Float16*)wsAlloc(p, (size_t)HIDF * UPF * 2);
  _Float16* wfT     = (_Float16*)wsAlloc(p, (size_t)HIDF * UPF * 2);
  _Float16* woT     = (_Float16*)wsAlloc(p, (size_t)HIDF * UPF * 2);
  _Float16* wdownT  = (_Float16*)wsAlloc(p, (size_t)INF * HIDF * 2);
  _Float16* wqT     = (_Float16*)wsAlloc(p, (size_t)NH * HSF * HSF * 2);
  _Float16* wkT     = (_Float16*)wsAlloc(p, (size_t)NH * HSF * HSF * 2);
  _Float16* wvT     = (_Float16*)wsAlloc(p, (size_t)NH * HSF * HSF * 2);
  float*    xul     = (float*)wsAlloc(p, (size_t)Bsz * UPF * 4);
  _Float16* xul_h   = (_Float16*)wsAlloc(p, (size_t)Bsz * UPF * 2);
  float*    xur     = (float*)wsAlloc(p, (size_t)Bsz * HIDF * 4);
  _Float16* xc_h    = (_Float16*)wsAlloc(p, (size_t)Bsz * UPF * 2);
  float*    x_skip  = (float*)wsAlloc(p, (size_t)Bsz * HIDF * 4);
  float*    qbuf    = (float*)wsAlloc(p, (size_t)Bsz * HIDF * 4);
  float*    kbuf    = (float*)wsAlloc(p, (size_t)Bsz * HIDF * 4);
  float*    vbuf    = (float*)wsAlloc(p, (size_t)Bsz * HIDF * 4);
  float*    it_buf  = (float*)wsAlloc(p, (size_t)Bsz * HIDF * 4);
  float*    ft_buf  = (float*)wsAlloc(p, (size_t)Bsz * HIDF * 4);
  float*    ot_buf  = (float*)wsAlloc(p, (size_t)Bsz * HIDF * 4);
  _Float16* qT_h    = (_Float16*)wsAlloc(p, (size_t)Bsz * HIDF * 2);
  _Float16* nT_h    = (_Float16*)wsAlloc(p, (size_t)Bsz * HIDF * 2);
  _Float16* pre_h   = (_Float16*)wsAlloc(p, (size_t)Bsz * HIDF * 2);
  float*    denom   = (float*)wsAlloc(p, (size_t)HIDF * 4);

  dim3 trb(32, 8);

  // 1) LayerNorm
  ln_k<<<Bsz, 256, 0, stream>>>(x, ln_g, ln_b, xn_h);

  // 2) weight transposes -> f16 (K-contiguous B^T for all GEMMs)
  tr_f32_f16_k<<<dim3(UPF/32,  INF/32, 1), trb, 0, stream>>>(w_ul,   wulT,   INF,  UPF,  0, 0);
  tr_f32_f16_k<<<dim3(HIDF/32, INF/32, 1), trb, 0, stream>>>(w_ur,   wurT,   INF,  HIDF, 0, 0);
  tr_f32_f16_k<<<dim3(HIDF/32, UPF/32, 1), trb, 0, stream>>>(w_skip, wskipT, UPF,  HIDF, 0, 0);
  tr_f32_f16_k<<<dim3(HIDF/32, UPF/32, 1), trb, 0, stream>>>(w_i,    wiT,    UPF,  HIDF, 0, 0);
  tr_f32_f16_k<<<dim3(HIDF/32, UPF/32, 1), trb, 0, stream>>>(w_f,    wfT,    UPF,  HIDF, 0, 0);
  tr_f32_f16_k<<<dim3(HIDF/32, UPF/32, 1), trb, 0, stream>>>(w_o,    woT,    UPF,  HIDF, 0, 0);
  tr_f32_f16_k<<<dim3(INF/32,  HIDF/32,1), trb, 0, stream>>>(w_down, wdownT, HIDF, INF,  0, 0);
  tr_f32_f16_k<<<dim3(HSF/32, HSF/32, NH), trb, 0, stream>>>(w_q, wqT, HSF, HSF, HSF*HSF, HSF*HSF);
  tr_f32_f16_k<<<dim3(HSF/32, HSF/32, NH), trb, 0, stream>>>(w_k, wkT, HSF, HSF, HSF*HSF, HSF*HSF);
  tr_f32_f16_k<<<dim3(HSF/32, HSF/32, NH), trb, 0, stream>>>(w_v, wvT, HSF, HSF, HSF*HSF, HSF*HSF);

  // 3) xul = xn@w_ul (+ f16 mirror), xur = xn@w_ur
  gemm_wmma_k<<<dim3(UPF/GTN, Bsz/GTM, 1), 256, 0, stream>>>(
      xn_h, INF, 0, wulT, INF, 0, xul, UPF, 0, b_ul, 0, nullptr, 0, xul_h, UPF, 1.f, INF);
  gemm_wmma_k<<<dim3(HIDF/GTN, Bsz/GTM, 1), 256, 0, stream>>>(
      xn_h, INF, 0, wurT, INF, 0, xur, HIDF, 0, b_ur, 0, nullptr, 0, nullptr, 0, 1.f, INF);

  // 4) xc = silu(conv(xul)) -> f16
  conv_silu_k<<<(size_t)Bsz * UPF / 256, 256, 0, stream>>>(xul, conv_w, conv_b, xc_h);

  // 5) dense projections from xc / xul
  gemm_wmma_k<<<dim3(HIDF/GTN, Bsz/GTM, 1), 256, 0, stream>>>(
      xc_h, UPF, 0, wskipT, UPF, 0, x_skip, HIDF, 0, b_skip, 0, nullptr, 0, nullptr, 0, 1.f, UPF);
  gemm_wmma_k<<<dim3(HIDF/GTN, Bsz/GTM, 1), 256, 0, stream>>>(
      xc_h, UPF, 0, wiT, UPF, 0, it_buf, HIDF, 0, b_i, 0, nullptr, 0, nullptr, 0, 1.f, UPF);
  gemm_wmma_k<<<dim3(HIDF/GTN, Bsz/GTM, 1), 256, 0, stream>>>(
      xc_h, UPF, 0, wfT, UPF, 0, ft_buf, HIDF, 0, b_f, 0, nullptr, 0, nullptr, 0, 1.f, UPF);
  gemm_wmma_k<<<dim3(HIDF/GTN, Bsz/GTM, 1), 256, 0, stream>>>(
      xul_h, UPF, 0, woT, UPF, 0, ot_buf, HIDF, 0, b_o, 0, nullptr, 0, nullptr, 0, 1.f, UPF);

  // 6) block-diagonal q/k/v: batched over heads (z), per-head 8192x256x256
  gemm_wmma_k<<<dim3(HSF/GTN, Bsz/GTM, NH), 256, 0, stream>>>(
      xc_h, UPF, HSF, wqT, HSF, HSF*HSF, qbuf, HIDF, HSF, b_q, HSF, nullptr, 0, nullptr, 0, 1.f, HSF);
  gemm_wmma_k<<<dim3(HSF/GTN, Bsz/GTM, NH), 256, 0, stream>>>(
      xc_h, UPF, HSF, wkT, HSF, HSF*HSF, kbuf, HIDF, HSF, b_k, HSF, nullptr, 0, nullptr, 0, 0.0625f, HSF);
  gemm_wmma_k<<<dim3(HSF/GTN, Bsz/GTM, NH), 256, 0, stream>>>(
      xul_h, UPF, HSF, wvT, HSF, HSF*HSF, vbuf, HIDF, HSF, b_v, HSF, nullptr, 0, nullptr, 0, 1.f, HSF);

  // 7) state update -> m_t, c_t, n_t (directly into d_out)
  state_k<<<(size_t)Bsz * HIDF / 256, 256, 0, stream>>>(
      it_buf, ft_buf, kbuf, vbuf, c_prev, n_prev, m_prev, out + o4, out + o2, out + o3);

  // 8) transposes for denom GEMM: qT, nT (f16 [HID x B])
  tr_f32_f16_k<<<dim3(HIDF/32, Bsz/32, 1), trb, 0, stream>>>(qbuf,     qT_h, Bsz, HIDF, 0, 0);
  tr_f32_f16_k<<<dim3(HIDF/32, Bsz/32, 1), trb, 0, stream>>>(out + o3, nT_h, Bsz, HIDF, 0, 0);

  // 9) denom = rowmax |n_t^T @ q|   (2048x2048x8192 WMMA reduction)
  denom_init_k<<<HIDF / 256, 256, 0, stream>>>(denom);
  denom_wmma_k<<<dim3(HIDF/DTN, HIDF/DTM, 1), 256, 0, stream>>>(nT_h, qT_h, denom, Bsz);

  // 10) h_t, GroupNorm, gated skip -> pre (f16); h_t into d_out
  head_out_k<<<dim3(Bsz, NH, 1), 256, 0, stream>>>(
      ot_buf, out + o2, qbuf, denom, gn_g, gn_b, x_skip, xur, out + o1, pre_h);

  // 11) out = pre @ w_down + b_down + x
  gemm_wmma_k<<<dim3(INF/GTN, Bsz/GTM, 1), 256, 0, stream>>>(
      pre_h, HIDF, 0, wdownT, HIDF, 0, out, INF, 0, b_down, 0, x, INF, nullptr, 0, 1.f, HIDF);
}
